// TinyYOLOv3_14405320311480
// MI455X (gfx1250) — compile-verified, matrix operand order checked by reading, not run on hardware
//
#include <hip/hip_runtime.h>
#include <cstdint>
#include <cstddef>

typedef __attribute__((ext_vector_type(16))) _Float16 v16h;
typedef __attribute__((ext_vector_type(8)))  float    v8f;

// ---------------------------------------------------------------------------
// Weight packing: HWIO f32 -> f16 WMMA B-fragment layout.
// B (K x N) fragment for wave32: lane l holds column n = nt*16 + (l&15);
// slot i holds k = kc*32 + ((i&8)<<1) + ((l>>4)<<3) + (i&7)   (ISA 7.12.2).
// HWIO flattened is exactly [k][n] with k = (dy*kw+dx)*ci + c.
// dst element index == ((nt*KC + kc)*32 + l)*16 + i.  NT is padded to a
// multiple of 4 (zero-filled) so the conv kernel can N-block by 64.
// ---------------------------------------------------------------------------
__global__ void pack_w_kernel(const float* __restrict__ w, _Float16* __restrict__ dst,
                              int K, int N, int KC, int NT) {
  long long total = (long long)NT * KC * 32 * 16;
  long long t = (long long)blockIdx.x * blockDim.x + threadIdx.x;
  if (t >= total) return;
  int i  = (int)(t & 15);
  int l  = (int)((t >> 4) & 31);
  long long r = t >> 9;
  int kc = (int)(r % KC);
  int nt = (int)(r / KC);
  int n = nt * 16 + (l & 15);
  int k = kc * 32 + ((i & 8) << 1) + ((l >> 4) << 3) + (i & 7);
  float v = 0.f;
  if (k < K && n < N) v = w[(long long)k * N + n];
  dst[t] = (_Float16)v;
}

// Fold BatchNorm (inference) into per-channel scale/bias. has_bn==0 -> head bias only.
__global__ void bn_fold_kernel(const float* __restrict__ g, const float* __restrict__ b,
                               const float* __restrict__ m, const float* __restrict__ v,
                               float* __restrict__ scale, float* __restrict__ bias,
                               int n, int has_bn) {
  int i = blockIdx.x * blockDim.x + threadIdx.x;
  if (i >= n) return;
  if (has_bn) {
    float s = g[i] * rsqrtf(v[i] + 1e-3f);
    scale[i] = s;
    bias[i]  = b[i] - m[i] * s;
  } else {
    scale[i] = 1.f;
    bias[i]  = b[i];
  }
}

// ---------------------------------------------------------------------------
// Implicit-GEMM conv (SAME, stride 1) + folded BN + optional LeakyReLU(0.1).
// 256-thread blocks = 8 independent waves; each wave computes a
// 16-pixel x 64-channel output tile: one shared A fragment per K-chunk feeds
// FOUR back-to-back v_wmma_f32_16x16x32_f16 (4 accumulators). This quarters
// activation traffic per FLOP vs a 16x16 tile; packed weights stay L2-resident.
// ---------------------------------------------------------------------------
__global__ void conv_wmma_kernel(const float* __restrict__ in, int B, int H, int W, int Cin,
                                 int kh, int kw, const _Float16* __restrict__ wpack,
                                 int KC, int Co, const float* __restrict__ scale,
                                 const float* __restrict__ bias, int leaky, int MT,
                                 float* __restrict__ out) {
  const int l    = threadIdx.x & 31;          // lane in wave
  const int wid  = threadIdx.x >> 5;          // wave in block
  const int mt   = blockIdx.x * 8 + wid;      // M tile (16 pixels)
  const int nt4  = blockIdx.y;                // group of 4 N tiles (64 channels)
  if (mt >= MT) return;                       // wave-uniform: EXEC stays all-ones

  const int K  = kh * kw * Cin;
  const int ph = (kh - 1) / 2, pw = (kw - 1) / 2;
  const long long Mtot = (long long)B * H * W;
  const long long mrow = (long long)mt * 16 + (l & 15);

  int b = 0, y = 0, x = 0;
  bool inb = (mrow < Mtot);
  if (inb) {
    long long p = mrow;
    b = (int)(p / ((long long)H * W));
    int rem = (int)(p - (long long)b * H * W);
    y = rem / W;
    x = rem - y * W;
  }
  const int laneBase = (l >> 4) << 3;   // lanes 16..31 hold K-offset +8
  const bool fast = ((Cin & 31) == 0);  // K-chunk never crosses a filter tap

  union { v8f v; float e[8]; } acc0, acc1, acc2, acc3;
#pragma unroll
  for (int r = 0; r < 8; ++r) { acc0.e[r] = 0.f; acc1.e[r] = 0.f; acc2.e[r] = 0.f; acc3.e[r] = 0.f; }

  // Base of this wave-lane's B fragments for N sub-tile 0 of this 64-wide group.
  const _Float16* wl = wpack + (((long long)nt4 * 4) * KC * 32 + l) * 16;
  const long long jstride = (long long)KC * 512;   // halves between N sub-tiles

  for (int kc = 0; kc < KC; ++kc) {
    union { v16h v; _Float16 e[16]; } a;
    if (fast) {
      int tap = (kc * 32) / Cin;
      int cb  = kc * 32 - tap * Cin;
      int dy = tap / kw, dx = tap - dy * kw;
      int iy = y + dy - ph, ix = x + dx - pw;
      if (inb && iy >= 0 && iy < H && ix >= 0 && ix < W) {
        const float* p = in + (((long long)b * H + iy) * W + ix) * Cin + cb + laneBase;
        float4 f0 = *(const float4*)(p);
        float4 f1 = *(const float4*)(p + 4);
        float4 f2 = *(const float4*)(p + 16);
        float4 f3 = *(const float4*)(p + 20);
        a.e[0]  = (_Float16)f0.x; a.e[1]  = (_Float16)f0.y;
        a.e[2]  = (_Float16)f0.z; a.e[3]  = (_Float16)f0.w;
        a.e[4]  = (_Float16)f1.x; a.e[5]  = (_Float16)f1.y;
        a.e[6]  = (_Float16)f1.z; a.e[7]  = (_Float16)f1.w;
        a.e[8]  = (_Float16)f2.x; a.e[9]  = (_Float16)f2.y;
        a.e[10] = (_Float16)f2.z; a.e[11] = (_Float16)f2.w;
        a.e[12] = (_Float16)f3.x; a.e[13] = (_Float16)f3.y;
        a.e[14] = (_Float16)f3.z; a.e[15] = (_Float16)f3.w;
      } else {
#pragma unroll
        for (int i = 0; i < 16; ++i) a.e[i] = (_Float16)0.f;
      }
    } else {
#pragma unroll
      for (int i = 0; i < 16; ++i) {
        int k = kc * 32 + ((i & 8) << 1) + laneBase + (i & 7);
        float fv = 0.f;
        if (inb && k < K) {
          int tap = k / Cin;
          int c   = k - tap * Cin;
          int dy = tap / kw, dx = tap - dy * kw;
          int iy = y + dy - ph, ix = x + dx - pw;
          if (iy >= 0 && iy < H && ix >= 0 && ix < W)
            fv = in[(((long long)b * H + iy) * W + ix) * Cin + c];
        }
        a.e[i] = (_Float16)fv;
      }
    }
    const _Float16* wk = wl + (long long)kc * 512;
    v16h b0 = *(const v16h*)(wk);
    v16h b1 = *(const v16h*)(wk + jstride);
    v16h b2 = *(const v16h*)(wk + 2 * jstride);
    v16h b3 = *(const v16h*)(wk + 3 * jstride);
    __builtin_prefetch(wk + 512, 0, 0);              // global_prefetch_b8 (next K-chunk)
    // Four independent WMMAs sharing the A fragment: no D->A/B hazard between
    // them, so they issue back-to-back into the matrix pipe.
    acc0.v = __builtin_amdgcn_wmma_f32_16x16x32_f16(false, a.v, false, b0, (short)0, acc0.v, false, false);
    acc1.v = __builtin_amdgcn_wmma_f32_16x16x32_f16(false, a.v, false, b1, (short)0, acc1.v, false, false);
    acc2.v = __builtin_amdgcn_wmma_f32_16x16x32_f16(false, a.v, false, b2, (short)0, acc2.v, false, false);
    acc3.v = __builtin_amdgcn_wmma_f32_16x16x32_f16(false, a.v, false, b3, (short)0, acc3.v, false, false);
  }

  // C/D layout: lane holds column n_sub = l&15; vgpr r holds row r + laneBase.
  float* accs[4] = {acc0.e, acc1.e, acc2.e, acc3.e};
#pragma unroll
  for (int j = 0; j < 4; ++j) {
    const int n = nt4 * 64 + j * 16 + (l & 15);
    if (n < Co) {
      float s  = scale[n];
      float t0 = bias[n];
#pragma unroll
      for (int r = 0; r < 8; ++r) {
        long long M2 = (long long)mt * 16 + r + laneBase;
        if (M2 < Mtot) {
          float f = accs[j][r] * s + t0;
          if (leaky) f = (f > 0.f) ? f : 0.1f * f;
          out[M2 * Co + n] = f;
        }
      }
    }
  }
}

// 2x2 stride-2 max pool, NHWC.
__global__ void maxpool2_kernel(const float* __restrict__ in, float* __restrict__ out,
                                int B, int H, int W, int C) {
  int Ho = H >> 1, Wo = W >> 1;
  long long total = (long long)B * Ho * Wo * C;
  long long t = (long long)blockIdx.x * blockDim.x + threadIdx.x;
  if (t >= total) return;
  int c = (int)(t % C);
  long long r = t / C;
  int xo = (int)(r % Wo); r /= Wo;
  int yo = (int)(r % Ho);
  int b  = (int)(r / Ho);
  const float* p = in + (((long long)b * H + yo * 2) * W + xo * 2) * C + c;
  long long rs = (long long)W * C;
  float v = fmaxf(fmaxf(p[0], p[C]), fmaxf(p[rs], p[rs + C]));
  out[t] = v;
}

// ZeroPad((0,1),(0,1)) + 2x2 stride-1 max pool on 13x13 (zeros participate in max).
__global__ void padpool_kernel(const float* __restrict__ in, float* __restrict__ out,
                               int B, int C) {
  long long total = (long long)B * 13 * 13 * C;
  long long t = (long long)blockIdx.x * blockDim.x + threadIdx.x;
  if (t >= total) return;
  int c = (int)(t % C);
  long long r = t / C;
  int x = (int)(r % 13); r /= 13;
  int y = (int)(r % 13);
  int b = (int)(r / 13);
  const float* base = in + (long long)b * 13 * 13 * C + c;
  float v00 = base[((long long)y * 13 + x) * C];
  float v01 = (x + 1 < 13) ? base[((long long)y * 13 + x + 1) * C] : 0.f;
  float v10 = (y + 1 < 13) ? base[((long long)(y + 1) * 13 + x) * C] : 0.f;
  float v11 = (x + 1 < 13 && y + 1 < 13) ? base[((long long)(y + 1) * 13 + x + 1) * C] : 0.f;
  out[t] = fmaxf(fmaxf(v00, v01), fmaxf(v10, v11));
}

// Nearest 2x upsample of (B,13,13,128) concat with root (B,26,26,256) -> (B,26,26,384).
__global__ void upcat_kernel(const float* __restrict__ up, const float* __restrict__ root,
                             float* __restrict__ out, int B) {
  long long total = (long long)B * 26 * 26 * 384;
  long long t = (long long)blockIdx.x * blockDim.x + threadIdx.x;
  if (t >= total) return;
  int c = (int)(t % 384);
  long long r = t / 384;
  int x = (int)(r % 26); r /= 26;
  int y = (int)(r % 26);
  int b = (int)(r / 26);
  float v;
  if (c < 128)
    v = up[(((long long)b * 13 + (y >> 1)) * 13 + (x >> 1)) * 128 + c];
  else
    v = root[(((long long)b * 26 + y) * 26 + x) * 256 + (c - 128)];
  out[t] = v;
}

__device__ __forceinline__ float sigmf(float x) { return 1.f / (1.f + expf(-x)); }

// YOLO anchor decode for both scales -> (B,1690,5) x1,y1,x2,y2,obj.
__global__ void decode_kernel(const float* __restrict__ y1, const float* __restrict__ y2,
                              float* __restrict__ out) {
  int t = blockIdx.x * blockDim.x + threadIdx.x;
  if (t >= 16 * 1690) return;
  int b = t / 1690;
  int r = t % 1690;
  const float* f;
  int g, col, row, a;
  float aw, ah;
  if (r < 338) {                       // scale 1: 13x13, anchors[2:]
    g = 13;
    int cell = r >> 1; a = r & 1;
    row = cell / 13;  col = cell % 13;
    f = y1 + (((long long)(b * 13 + row) * 13 + col) * 10 + a * 5);
    aw = (a == 0) ? 0.23f : 0.42f;
    ah = (a == 0) ? 0.61f : 0.88f;
  } else {                             // scale 2: 26x26, anchors[:2]
    int rr = r - 338;
    g = 26;
    int cell = rr >> 1; a = rr & 1;
    row = cell / 26;  col = cell % 26;
    f = y2 + (((long long)(b * 26 + row) * 26 + col) * 10 + a * 5);
    aw = (a == 0) ? 0.08f : 0.14f;
    ah = (a == 0) ? 0.22f : 0.42f;
  }
  float cx = (sigmf(f[0]) + (float)col) / (float)g;
  float cy = (sigmf(f[1]) + (float)row) / (float)g;
  float w  = expf(f[2]) * aw;
  float h  = expf(f[3]) * ah;
  float ob = sigmf(f[4]);
  float* o = out + (long long)t * 5;
  o[0] = cx - 0.5f * w;
  o[1] = cy - 0.5f * h;
  o[2] = cx + 0.5f * w;
  o[3] = cy + 0.5f * h;
  o[4] = ob;
}

// ---------------------------------------------------------------------------
// Host launcher
// d_in flattening (setup_inputs dict order, nested dicts in insertion order):
//   [0]=x ; b1..b9: {w,g,b,m,v} ; b10:{w,b} ; b11:{w,g,b,m,v} ; b12:{...} ; b13:{w,b}
// ---------------------------------------------------------------------------
extern "C" void kernel_launch(void* const* d_in, const int* in_sizes, int n_in,
                              void* d_out, int out_size, void* d_ws, size_t ws_size,
                              hipStream_t stream) {
  (void)in_sizes; (void)n_in; (void)out_size; (void)ws_size;
  char* ws = (char*)d_ws;
  const int BATCH = 16;

  static const int khs[13]    = {3, 3, 3, 3, 3, 3, 3, 1, 3, 1, 1, 3, 1};
  static const int cis[13]    = {3, 16, 32, 64, 128, 256, 512, 1024, 256, 512, 256, 384, 256};
  static const int cos_[13]   = {16, 32, 64, 128, 256, 512, 1024, 256, 512, 10, 128, 256, 10};
  static const int ishead[13] = {0, 0, 0, 0, 0, 0, 0, 0, 0, 1, 0, 0, 1};

  int widx[13];
  { int idx = 1; for (int L = 0; L < 13; ++L) { widx[L] = idx; idx += ishead[L] ? 2 : 5; } }

  size_t woff[13], bnoff[13];
  int KCs[13], NTs[13], NT4s[13];
  size_t wcur = 0, bncur = (size_t)28 << 20;
  for (int L = 0; L < 13; ++L) {
    int K = khs[L] * khs[L] * cis[L];
    KCs[L]  = (K + 31) / 32;
    NT4s[L] = (cos_[L] + 63) / 64;        // 64-wide N groups per conv block.y
    NTs[L]  = NT4s[L] * 4;                // padded 16-wide tiles (zero-filled)
    woff[L] = wcur;
    size_t bytes = (size_t)NTs[L] * KCs[L] * 512 * sizeof(_Float16);
    wcur = (wcur + bytes + 255) & ~(size_t)255;
    bnoff[L] = bncur;
    bncur = (bncur + (size_t)2 * cos_[L] * sizeof(float) + 255) & ~(size_t)255;
  }

  // Per-launch weight pack + BN fold (deterministic, L2-resident afterwards).
  for (int L = 0; L < 13; ++L) {
    const float* w = (const float*)d_in[widx[L]];
    int K = khs[L] * khs[L] * cis[L];
    long long tot = (long long)NTs[L] * KCs[L] * 512;
    pack_w_kernel<<<dim3((unsigned)((tot + 255) / 256)), 256, 0, stream>>>(
        w, (_Float16*)(ws + woff[L]), K, cos_[L], KCs[L], NTs[L]);
    float* sc = (float*)(ws + bnoff[L]);
    float* bi = sc + cos_[L];
    unsigned bng = (unsigned)((cos_[L] + 255) / 256);
    if (ishead[L]) {
      const float* bb = (const float*)d_in[widx[L] + 1];
      bn_fold_kernel<<<dim3(bng), 256, 0, stream>>>(bb, bb, bb, bb, sc, bi, cos_[L], 0);
    } else {
      bn_fold_kernel<<<dim3(bng), 256, 0, stream>>>(
          (const float*)d_in[widx[L] + 1], (const float*)d_in[widx[L] + 2],
          (const float*)d_in[widx[L] + 3], (const float*)d_in[widx[L] + 4],
          sc, bi, cos_[L], 1);
    }
  }

  float* BUF1   = (float*)(ws + ((size_t)32  << 20));  // up to 177 MB (layer-1 output)
  float* BUF2   = (float*)(ws + ((size_t)216 << 20));  // up to 89 MB
  float* ROOT   = (float*)(ws + ((size_t)308 << 20));  // (16,26,26,256)
  float* BRANCH = (float*)(ws + ((size_t)321 << 20));  // (16,13,13,256)
  float* Y1     = (float*)(ws + ((size_t)324 << 20));  // (16,13,13,10)
  float* Y2     = (float*)(ws + ((size_t)326 << 20));  // (16,26,26,10)

  auto conv = [&](int L, const float* in, float* out, int H, int leaky) {
    long long M = (long long)BATCH * H * H;
    int MT = (int)((M + 15) / 16);
    dim3 grid((unsigned)((MT + 7) / 8), (unsigned)NT4s[L]);
    conv_wmma_kernel<<<grid, 256, 0, stream>>>(
        in, BATCH, H, H, cis[L], khs[L], khs[L],
        (const _Float16*)(ws + woff[L]), KCs[L], cos_[L],
        (const float*)(ws + bnoff[L]), (const float*)(ws + bnoff[L]) + cos_[L],
        leaky, MT, out);
  };
  auto pool = [&](const float* in, float* out, int H, int C) {
    long long tot = (long long)BATCH * (H / 2) * (H / 2) * C;
    maxpool2_kernel<<<dim3((unsigned)((tot + 255) / 256)), 256, 0, stream>>>(
        in, out, BATCH, H, H, C);
  };

  const float* x = (const float*)d_in[0];

  conv(0, x, BUF1, 416, 1);         pool(BUF1, BUF2, 416, 16);    // b1 -> 208
  conv(1, BUF2, BUF1, 208, 1);      pool(BUF1, BUF2, 208, 32);    // b2 -> 104
  conv(2, BUF2, BUF1, 104, 1);      pool(BUF1, BUF2, 104, 64);    // b3 -> 52
  conv(3, BUF2, BUF1, 52, 1);       pool(BUF1, BUF2, 52, 128);    // b4 -> 26
  conv(4, BUF2, ROOT, 26, 1);       pool(ROOT, BUF1, 26, 256);    // b5 (root) -> 13
  conv(5, BUF1, BUF2, 13, 1);                                     // b6 (13,512)
  {
    long long tot = (long long)BATCH * 13 * 13 * 512;
    padpool_kernel<<<dim3((unsigned)((tot + 255) / 256)), 256, 0, stream>>>(
        BUF2, BUF1, BATCH, 512);                                  // pad + stride-1 pool
  }
  conv(6, BUF1, BUF2, 13, 1);                                     // b7 (13,1024)
  conv(7, BUF2, BRANCH, 13, 1);                                   // b8 (13,256)
  conv(8, BRANCH, BUF1, 13, 1);                                   // b9 (13,512)
  conv(9, BUF1, Y1, 13, 0);                                       // b10 head
  conv(10, BRANCH, BUF1, 13, 1);                                  // b11 (13,128)
  {
    long long tot = (long long)BATCH * 26 * 26 * 384;
    upcat_kernel<<<dim3((unsigned)((tot + 255) / 256)), 256, 0, stream>>>(
        BUF1, ROOT, BUF2, BATCH);                                 // upsample + concat
  }
  conv(11, BUF2, BUF1, 26, 1);                                    // b12 (26,256)
  conv(12, BUF1, Y2, 26, 0);                                      // b13 head

  decode_kernel<<<dim3((16 * 1690 + 255) / 256), 256, 0, stream>>>(
      Y1, Y2, (float*)d_out);
}
